// Q6ArithmeticLayer_34359739039
// MI455X (gfx1250) — matrix operand order repaired and checked by reading, not compiled, not run on hardware
//
#include <hip/hip_runtime.h>

typedef __attribute__((ext_vector_type(16))) _Float16 v16h;
typedef __attribute__((ext_vector_type(8)))  float    v8f;

#define DD      1024          // feature dim
#define KITERS  (DD / 32)     // 32 WMMA k-steps
#define NPROTO  8
#define NK      6
#define WAVES   8             // waves per block (wave32)

__global__ __launch_bounds__(WAVES * 32)
void q6_route_kernel(const float* __restrict__ x,
                     const float* __restrict__ W,
                     const float* __restrict__ protos,
                     const float* __restrict__ hs_ptr,
                     float* __restrict__ out,
                     int ntok)
{
    // B fragments (W^T, f16, zero-padded cols 6..15) in contiguous k order (k = 16*hi + e, same as A).
    // 32 k-steps * 32 lanes * 32B = 32 KB LDS, shared by all 8 waves.
    __shared__ v16h bfrag[KITERS][32];

    const int lane = threadIdx.x & 31;
    const int wave = threadIdx.x >> 5;
    const int n    = lane & 15;   // B column / A row within tile / C column
    const int hi   = lane >> 4;   // half-wave selector

    for (int idx = threadIdx.x; idx < KITERS * 32; idx += blockDim.x) {
        const int kit = idx >> 5;
        const int ln  = idx & 31;
        const int bn  = ln & 15;
        const int bhi = ln >> 4;
        v16h v;
        if (bn < NK) {
            // B[k][n] = W[n][k]; lane ln element e holds K = kit*32 + bhi*16 + e
            const float* wp = W + bn * DD + kit * 32 + bhi * 16;
            #pragma unroll
            for (int e = 0; e < 16; ++e) v[e] = (_Float16)wp[e];
        } else {
            #pragma unroll
            for (int e = 0; e < 16; ++e) v[e] = (_Float16)0.0f;
        }
        bfrag[kit][ln] = v;
    }
    __syncthreads();

    const int t0 = (blockIdx.x * WAVES + wave) * 16;   // first token of this wave's tile
    if (t0 >= ntok) return;

    // A row for this lane: token t0 + (lane&15)
    const float* xrow = x + (size_t)(t0 + n) * DD;

    v8f c = {0.f, 0.f, 0.f, 0.f, 0.f, 0.f, 0.f, 0.f};

    #pragma unroll 4
    for (int kit = 0; kit < KITERS; ++kit) {
        const int k0 = kit * 32;
        // A operand, contiguous k order: elems 0..15 -> K=[k0+16*hi, +16) (matches B staging)
        const float4* p0 = (const float4*)(xrow + k0 + hi * 16);
        const float4* p1 = (const float4*)(xrow + k0 + hi * 16 + 8);
        const float4 f0 = p0[0], f1 = p0[1];
        const float4 f2 = p1[0], f3 = p1[1];
        v16h a;
        a[0]  = (_Float16)f0.x; a[1]  = (_Float16)f0.y; a[2]  = (_Float16)f0.z; a[3]  = (_Float16)f0.w;
        a[4]  = (_Float16)f1.x; a[5]  = (_Float16)f1.y; a[6]  = (_Float16)f1.z; a[7]  = (_Float16)f1.w;
        a[8]  = (_Float16)f2.x; a[9]  = (_Float16)f2.y; a[10] = (_Float16)f2.z; a[11] = (_Float16)f2.w;
        a[12] = (_Float16)f3.x; a[13] = (_Float16)f3.y; a[14] = (_Float16)f3.z; a[15] = (_Float16)f3.w;

        const v16h b = bfrag[kit][lane];
        c = __builtin_amdgcn_wmma_f32_16x16x32_f16(
                /*neg_a=*/false, a, /*neg_b=*/false, b,
                /*c_mod=*/(short)0, c, /*reuse_a=*/false, /*reuse_b=*/false);
    }

    const float hs = hs_ptr[0];

    // Each of lanes n<8 owns prototype n (L2-normalized, eps clamp 1e-12).
    float pr[NK] = {0.f, 0.f, 0.f, 0.f, 0.f, 0.f};
    if (n < NPROTO) {
        float s = 0.f;
        #pragma unroll
        for (int j = 0; j < NK; ++j) { const float v = protos[n * NK + j]; s += v * v; }
        const float norm = fmaxf(sqrtf(s), 1e-12f);
        #pragma unroll
        for (int j = 0; j < NK; ++j) pr[j] = protos[n * NK + j] / norm;
    }

    // C/D layout: VGPR r holds row M = r + 8*hi, column N = n.
    #pragma unroll
    for (int r = 0; r < 8; ++r) {
        const float z = tanhf(c[r]);          // cols n>=6 accumulated exact 0 -> z = 0
        // L2 norm across the 6 (16) columns: reduce over the 16-lane half.
        float ss = z * z;
        ss += __shfl_xor(ss, 1, 32);
        ss += __shfl_xor(ss, 2, 32);
        ss += __shfl_xor(ss, 4, 32);
        ss += __shfl_xor(ss, 8, 32);
        const float zn = z / fmaxf(sqrtf(ss), 1e-6f);

        // dot with this lane's prototype: broadcast z[0..5] of this row.
        float dot = 0.f;
        #pragma unroll
        for (int j = 0; j < NK; ++j) {
            const float zj = __shfl(zn, hi * 16 + j, 32);
            dot += zj * pr[j];
        }

        // dist = (6 - 6*dot)*0.5 ; logits = -hs*dist ; stable softmax over 8 protos
        const float logit = -hs * 0.5f * (6.0f - 6.0f * dot);
        float mx = logit;
        mx = fmaxf(mx, __shfl_xor(mx, 1, 32));
        mx = fmaxf(mx, __shfl_xor(mx, 2, 32));
        mx = fmaxf(mx, __shfl_xor(mx, 4, 32));
        const float ev = __expf(logit - mx);
        float se = ev;
        se += __shfl_xor(se, 1, 32);
        se += __shfl_xor(se, 2, 32);
        se += __shfl_xor(se, 4, 32);

        if (n < NPROTO) {
            const int token = t0 + r + 8 * hi;
            out[(size_t)token * NPROTO + n] = ev / se;
        }
    }
}

extern "C" void kernel_launch(void* const* d_in, const int* in_sizes, int n_in,
                              void* d_out, int out_size, void* d_ws, size_t ws_size,
                              hipStream_t stream) {
    const float* x  = (const float*)d_in[0];   // (B,T,D) f32
    const float* W  = (const float*)d_in[1];   // (6,D)   f32
    const float* pr = (const float*)d_in[2];   // (8,6)   f32
    const float* hs = (const float*)d_in[3];   // scalar  f32
    float* out = (float*)d_out;                // (B,T,8) f32

    const int ntok = in_sizes[0] / DD;                 // B*T = 32768
    const int tokensPerBlock = WAVES * 16;             // 128
    const int blocks = (ntok + tokensPerBlock - 1) / tokensPerBlock;  // 256

    q6_route_kernel<<<blocks, WAVES * 32, 0, stream>>>(x, W, pr, hs, out, ntok);
}
